// QMoELayer_25786983645542
// MI455X (gfx1250) — matrix operand order repaired and checked by reading, not compiled, not run on hardware
//
#include <hip/hip_runtime.h>
#include <hip/hip_bf16.h>

// ---------------------------------------------------------------------------
// MoE transformer encoder layer (top-2 of 4 experts) for MI455X / gfx1250.
// GEMMs: v_wmma_f32_16x16x32_f16 (f16 in, f32 accumulate).
// Weight-layout GEMMs stage the shared B tile in LDS via
// global_load_async_to_lds_b128 (ASYNCcnt) to kill the 8x redundant
// per-wave global B traffic. Softmax / LayerNorm / gating stay fp32.
// ---------------------------------------------------------------------------

typedef _Float16 half_t;
typedef __attribute__((ext_vector_type(16))) _Float16 v16h;
typedef __attribute__((ext_vector_type(8)))  _Float16 v8h;
typedef __attribute__((ext_vector_type(8)))  float    v8f;

#define BQ   8
#define SQ   512
#define DQ   1024
#define HQ   16
#define DFFQ 4096
#define EQ   4
#define EPSQ 1e-5f

// ---------------------------------------------------------------------------
// Weight-layout WMMA GEMM: C[M,N] = act( scale*(A[M,K] @ W[N,K]^T) + bias[N] )
// Block = 256 threads = 8 waves; block tile 128x64; wave tile 16x64.
// B tile (64 rows x 64 halfs, 8KB) staged in LDS with async copies per
// K-round of 64; requires K % 64 == 0 (holds for all launches here).
// ---------------------------------------------------------------------------
__global__ __launch_bounds__(256) void gemm_wmma_nk_kernel(
    const half_t* __restrict__ A, long lda, long sAb, long sAh,
    const half_t* __restrict__ Bm, long ldb, long sBb, long sBh,
    float* __restrict__ Cf, half_t* __restrict__ Ch, long ldc, long sCb, long sCh,
    const float* __restrict__ bias, float scale, int relu,
    int K, int zH, int rows_per_b,
    const float* __restrict__ gate, int expert)
{
    int b, h = 0;
    if (gridDim.z > 1) { b = (int)(blockIdx.z / (unsigned)zH); h = (int)(blockIdx.z % (unsigned)zH); }
    else               { b = (int)((blockIdx.x * 128u) / (unsigned)rows_per_b); }
    if (gate[b * EQ + expert] == 0.0f) return;   // expert not routed to this batch item

    __shared__ __align__(32) half_t btile[64 * 64];   // [n_local][k_local], 8KB

    const int tid  = threadIdx.x;
    const int wave = tid >> 5;
    const int lane = tid & 31;
    const int hs   = lane >> 4;      // half-wave select (wave32)
    const int l16  = lane & 15;

    const long m0 = (long)blockIdx.x * 128 + wave * 16;
    const long n0 = (long)blockIdx.y * 64;

    const half_t* Arow  = A + (long)b * sAb + (long)h * sAh + (m0 + l16) * lda;
    const half_t* Brow0 = Bm + (long)b * sBb + (long)h * sBh + n0 * ldb;

    // Async-copy assignment: 4 threads per B row, 2x16B chunks per thread.
    const int crow = tid >> 2;        // 0..63  (n_local)
    const int cchk = tid & 3;         // chunk pair (c, c+4) of 16B within 128B row
    const uint32_t ldsbase = (uint32_t)(uintptr_t)(&btile[0]);
    const uint32_t l0 = ldsbase + (uint32_t)(crow * 128 + cchk * 16);

    v8f acc[4];
#pragma unroll
    for (int t = 0; t < 4; ++t) {
        v8f z = {0.f, 0.f, 0.f, 0.f, 0.f, 0.f, 0.f, 0.f};
        acc[t] = z;
    }

    for (long k0 = 0; k0 < (long)K; k0 += 64) {
        __syncthreads();              // previous round's LDS reads complete
        {
            const half_t* g0 = Brow0 + (long)crow * ldb + k0 + cchk * 8;
            const half_t* g1 = g0 + 32;                     // chunk c+4
            asm volatile("global_load_async_to_lds_b128 %0, %1, off"
                         :: "v"(l0), "v"((unsigned long long)(uintptr_t)g0)
                         : "memory");
            asm volatile("global_load_async_to_lds_b128 %0, %1, off"
                         :: "v"((uint32_t)(l0 + 64)), "v"((unsigned long long)(uintptr_t)g1)
                         : "memory");
        }
        if (k0 + 64 < (long)K)
            __builtin_prefetch((const void*)(Arow + k0 + 64), 0, 1);  // global_prefetch_b8
        asm volatile("s_wait_asynccnt 0x0" ::: "memory");
        __syncthreads();              // whole block sees the staged tile

#pragma unroll
        for (int kk = 0; kk < 64; kk += 32) {
            // A fragment, contiguous k order (same order as B below):
            //  lanes 0-15 : M=lane,    elems 0-15 = K..K+15
            //  lanes16-31 : M=lane-16, elems 0-15 = K+16..K+31
            v16h a;
            {
                const half_t* ap = Arow + k0 + kk + hs * 16;
                v8h p0 = *(const v8h*)(ap);
                v8h p1 = *(const v8h*)(ap + 8);
#pragma unroll
                for (int i = 0; i < 8; ++i) { a[i] = p0[i]; a[8 + i] = p1[i]; }
            }
#pragma unroll
            for (int t = 0; t < 4; ++t) {
                // B fragment from LDS: lanes 0-15 b[i]=B[kk+i][n], 16-31 b[i]=B[kk+16+i][n]
                const v16h bf = *(const v16h*)(btile + (t * 16 + l16) * 64 + kk + hs * 16);
                acc[t] = __builtin_amdgcn_wmma_f32_16x16x32_f16(
                    false, a, false, bf, (short)0, acc[t], false, false);
            }
        }
    }

    // Epilogue. C/D layout: vgpr r -> row m0 + 8*hs + r, col n0 + 16*t + l16.
#pragma unroll
    for (int t = 0; t < 4; ++t) {
        const long col = n0 + t * 16 + l16;
        const float bv = bias ? bias[col] : 0.0f;
#pragma unroll
        for (int r = 0; r < 8; ++r) {
            const long row = m0 + hs * 8 + r;
            float v = acc[t][r] * scale + bv;
            if (relu) v = v > 0.0f ? v : 0.0f;
            const long off = (long)b * sCb + (long)h * sCh + row * ldc + col;
            if (Cf) Cf[off] = v;
            if (Ch) Ch[off] = (half_t)v;
        }
    }
}

// ---------------------------------------------------------------------------
// Value-layout WMMA GEMM (B is f16 [K,N] row-major): attention P@V only.
// ---------------------------------------------------------------------------
__global__ __launch_bounds__(256) void gemm_wmma_kn_kernel(
    const half_t* __restrict__ A, long lda, long sAb, long sAh,
    const half_t* __restrict__ Bm, long ldb, long sBb, long sBh,
    half_t* __restrict__ Ch, long ldc, long sCb, long sCh,
    float scale, int K, int zH,
    const float* __restrict__ gate, int expert)
{
    const int b = (int)(blockIdx.z / (unsigned)zH);
    const int h = (int)(blockIdx.z % (unsigned)zH);
    if (gate[b * EQ + expert] == 0.0f) return;

    const int tid  = threadIdx.x;
    const int wave = tid >> 5;
    const int lane = tid & 31;
    const int hs   = lane >> 4;
    const int l16  = lane & 15;

    const long m0 = (long)blockIdx.x * 128 + wave * 16;
    const long n0 = (long)blockIdx.y * 64;

    const half_t* Arow  = A + (long)b * sAb + (long)h * sAh + (m0 + l16) * lda;
    const half_t* Bbase = Bm + (long)b * sBb + (long)h * sBh;

    v8f acc[4];
#pragma unroll
    for (int t = 0; t < 4; ++t) {
        v8f z = {0.f, 0.f, 0.f, 0.f, 0.f, 0.f, 0.f, 0.f};
        acc[t] = z;
    }

    for (long k = 0; k < (long)K; k += 32) {
        if (k + 32 < (long)K)
            __builtin_prefetch((const void*)(Arow + k + 32), 0, 1);
        v16h a;
        {
            const half_t* ap = Arow + k + hs * 16;
            v8h p0 = *(const v8h*)(ap);
            v8h p1 = *(const v8h*)(ap + 8);
#pragma unroll
            for (int i = 0; i < 8; ++i) { a[i] = p0[i]; a[8 + i] = p1[i]; }
        }
#pragma unroll
        for (int t = 0; t < 4; ++t) {
            v16h bf;
            const half_t* bp = Bbase + (k + hs * 16) * ldb + n0 + t * 16 + l16;
#pragma unroll
            for (int i = 0; i < 16; ++i) bf[i] = bp[(long)i * ldb];
            acc[t] = __builtin_amdgcn_wmma_f32_16x16x32_f16(
                false, a, false, bf, (short)0, acc[t], false, false);
        }
    }

#pragma unroll
    for (int t = 0; t < 4; ++t) {
        const long col = n0 + t * 16 + l16;
#pragma unroll
        for (int r = 0; r < 8; ++r) {
            const long row = m0 + hs * 8 + r;
            const long off = (long)b * sCb + (long)h * sCh + row * ldc + col;
            Ch[off] = (half_t)(acc[t][r] * scale);
        }
    }
}

// ---------------------------------------------------------------------------
// Router: logits[b,e] = mean_s(x[b]) . gate_w[e] + gate_b[e]; top-2 mask *
// softmax(all logits) -> w[b*E+e] (0 for unselected experts).
// ---------------------------------------------------------------------------
__global__ __launch_bounds__(256) void gate_kernel(
    const float* __restrict__ x, const float* __restrict__ gw,
    const float* __restrict__ gb, float* __restrict__ wout)
{
    __shared__ float sh[256];
    __shared__ float logits[EQ];
    const int b = blockIdx.x, tid = threadIdx.x;
    const float* xb = x + (long)b * SQ * DQ;
    float acc[EQ] = {0.f, 0.f, 0.f, 0.f};
    for (int i = tid; i < SQ * DQ; i += 256) {
        const float xv = xb[i];
        const int d = i & (DQ - 1);
#pragma unroll
        for (int e = 0; e < EQ; ++e) acc[e] += xv * gw[e * DQ + d];
    }
    for (int e = 0; e < EQ; ++e) {
        sh[tid] = acc[e];
        __syncthreads();
        for (int s = 128; s > 0; s >>= 1) {
            if (tid < s) sh[tid] += sh[tid + s];
            __syncthreads();
        }
        if (tid == 0) logits[e] = sh[0] / (float)SQ + gb[e];
        __syncthreads();
    }
    if (tid == 0) {
        int i1 = 0;
        for (int e = 1; e < EQ; ++e) if (logits[e] > logits[i1]) i1 = e;
        int i2 = -1;
        for (int e = 0; e < EQ; ++e)
            if (e != i1 && (i2 < 0 || logits[e] > logits[i2])) i2 = e;
        float mx = logits[0];
        for (int e = 1; e < EQ; ++e) mx = fmaxf(mx, logits[e]);
        float p[EQ], s = 0.f;
        for (int e = 0; e < EQ; ++e) { p[e] = __expf(logits[e] - mx); s += p[e]; }
        for (int e = 0; e < EQ; ++e)
            wout[b * EQ + e] = (e == i1 || e == i2) ? p[e] / s : 0.0f;
    }
}

// Row softmax over 512 f16 scores, in place. One block per (b,h,q) row.
__global__ __launch_bounds__(256) void softmax_rows_kernel(
    half_t* __restrict__ p, const float* __restrict__ gate, int expert)
{
    const long row = blockIdx.x;
    const int b = (int)(row / ((long)HQ * SQ));
    if (gate[b * EQ + expert] == 0.0f) return;
    __shared__ float sh[256];
    const int tid = threadIdx.x;
    half_t* r = p + row * SQ;
    const float v0 = (float)r[tid], v1 = (float)r[tid + 256];
    sh[tid] = fmaxf(v0, v1);
    __syncthreads();
    for (int s = 128; s > 0; s >>= 1) {
        if (tid < s) sh[tid] = fmaxf(sh[tid], sh[tid + s]);
        __syncthreads();
    }
    const float m = sh[0];
    __syncthreads();
    const float e0 = __expf(v0 - m), e1 = __expf(v1 - m);
    sh[tid] = e0 + e1;
    __syncthreads();
    for (int s = 128; s > 0; s >>= 1) {
        if (tid < s) sh[tid] += sh[tid + s];
        __syncthreads();
    }
    const float inv = 1.0f / sh[0];
    r[tid] = (half_t)(e0 * inv);
    r[tid + 256] = (half_t)(e1 * inv);
}

// Fused residual-add + LayerNorm over D=1024; writes f32 (+ optional f16).
__global__ __launch_bounds__(256) void add_ln_kernel(
    const float* __restrict__ resid, const float* __restrict__ y,
    const float* __restrict__ w, const float* __restrict__ bb,
    float* __restrict__ outf, half_t* __restrict__ outh,
    const float* __restrict__ gate, int expert)
{
    const long row = blockIdx.x;                 // over B*S
    const int b = (int)(row / SQ);
    if (gate[b * EQ + expert] == 0.0f) return;
    __shared__ float sh[256];
    const int tid = threadIdx.x;
    const long base = row * DQ;
    float t[4];
#pragma unroll
    for (int i = 0; i < 4; ++i) {
        const long idx = base + tid + i * 256;
        t[i] = y[idx] + (resid ? resid[idx] : 0.0f);
    }
    sh[tid] = t[0] + t[1] + t[2] + t[3];
    __syncthreads();
    for (int s = 128; s > 0; s >>= 1) {
        if (tid < s) sh[tid] += sh[tid + s];
        __syncthreads();
    }
    const float mean = sh[0] / (float)DQ;
    __syncthreads();
    float vs = 0.f;
#pragma unroll
    for (int i = 0; i < 4; ++i) { const float d = t[i] - mean; vs += d * d; }
    sh[tid] = vs;
    __syncthreads();
    for (int s = 128; s > 0; s >>= 1) {
        if (tid < s) sh[tid] += sh[tid + s];
        __syncthreads();
    }
    const float rstd = rsqrtf(sh[0] / (float)DQ + EPSQ);
#pragma unroll
    for (int i = 0; i < 4; ++i) {
        const int d = tid + i * 256;
        const float o = (t[i] - mean) * rstd * w[d] + bb[d];
        outf[base + d] = o;
        if (outh) outh[base + d] = (half_t)o;
    }
}

__global__ void zero_f32_kernel(float* __restrict__ p, long n)
{
    for (long i = blockIdx.x * 256L + threadIdx.x; i < n; i += (long)gridDim.x * 256L)
        p[i] = 0.0f;
}

__global__ void cast_f16_kernel(const float* __restrict__ in, half_t* __restrict__ out, long n)
{
    for (long i = blockIdx.x * 256L + threadIdx.x; i < n; i += (long)gridDim.x * 256L)
        out[i] = (half_t)in[i];
}

__global__ void accum_kernel(float* __restrict__ out, const float* __restrict__ y,
                             const float* __restrict__ gate, int expert, long n)
{
    for (long i = blockIdx.x * 256L + threadIdx.x; i < n; i += (long)gridDim.x * 256L) {
        const int b = (int)(i / ((long)SQ * DQ));
        out[i] += gate[b * EQ + expert] * y[i];   // w==0 -> contributes exactly 0
    }
}

// ---------------------------------------------------------------------------
static void launch_gemm_nk(hipStream_t stream,
                           const half_t* A, long lda, long sAb, long sAh,
                           const half_t* Bm, long ldb, long sBb, long sBh,
                           float* Cf, half_t* Ch, long ldc, long sCb, long sCh,
                           const float* bias, float scale, int relu,
                           long M, long N, int K, int Z, int zH, int rows_per_b,
                           const float* gate, int expert)
{
    dim3 grid((unsigned)(M / 128), (unsigned)(N / 64), (unsigned)Z);
    gemm_wmma_nk_kernel<<<grid, 256, 0, stream>>>(
        A, lda, sAb, sAh, Bm, ldb, sBb, sBh, Cf, Ch, ldc, sCb, sCh,
        bias, scale, relu, K, zH, rows_per_b, gate, expert);
}

extern "C" void kernel_launch(void* const* d_in, const int* in_sizes, int n_in,
                              void* d_out, int out_size, void* d_ws, size_t ws_size,
                              hipStream_t stream)
{
    (void)in_sizes; (void)n_in; (void)out_size; (void)ws_size;
    const float* x      = (const float*)d_in[0];
    const float* in_w   = (const float*)d_in[1];
    const float* in_b   = (const float*)d_in[2];
    const float* out_w  = (const float*)d_in[3];
    const float* out_b  = (const float*)d_in[4];
    const float* ln1_w  = (const float*)d_in[5];
    const float* ln1_b  = (const float*)d_in[6];
    const float* lin1_w = (const float*)d_in[7];
    const float* lin1_b = (const float*)d_in[8];
    const float* lin2_w = (const float*)d_in[9];
    const float* lin2_b = (const float*)d_in[10];
    const float* ln2_w  = (const float*)d_in[11];
    const float* ln2_b  = (const float*)d_in[12];
    const float* gate_w = (const float*)d_in[13];
    const float* gate_b = (const float*)d_in[14];
    float* out = (float*)d_out;

    const long BSD  = (long)BQ * SQ * DQ;          // 4 Mi elems
    const long S3D  = 3L * DQ;                     // 3072
    char* ws = (char*)d_ws;
    size_t off = 0;
    auto alloc = [&](size_t bytes) -> void* {
        void* p = (void*)(ws + off);
        off += (bytes + 255) & ~(size_t)255;
        return p;
    };

    float*  wbuf   = (float*)alloc((size_t)BQ * EQ * 4);
    half_t* xh     = (half_t*)alloc((size_t)BSD * 2);
    half_t* iph    = (half_t*)alloc((size_t)EQ * 3 * DQ * DQ * 2);
    half_t* oph    = (half_t*)alloc((size_t)EQ * DQ * DQ * 2);
    half_t* l1h    = (half_t*)alloc((size_t)EQ * DFFQ * DQ * 2);
    half_t* l2h    = (half_t*)alloc((size_t)EQ * DQ * DFFQ * 2);
    half_t* qkvh   = (half_t*)alloc((size_t)BQ * SQ * S3D * 2);
    half_t* sbuf   = (half_t*)alloc((size_t)BQ * HQ * SQ * SQ * 2);   // scores -> probs
    half_t* oh     = (half_t*)alloc((size_t)BSD * 2);
    float*  oprojf = (float*)alloc((size_t)BSD * 4);
    float*  ln1f   = (float*)alloc((size_t)BSD * 4);
    half_t* ln1h   = (half_t*)alloc((size_t)BSD * 2);
    half_t* ffh    = (half_t*)alloc((size_t)BQ * SQ * DFFQ * 2);
    float*  ff2f   = (float*)alloc((size_t)BSD * 4);
    float*  ln2f   = (float*)alloc((size_t)BSD * 4);

    zero_f32_kernel<<<4096, 256, 0, stream>>>(out, BSD);
    gate_kernel<<<BQ, 256, 0, stream>>>(x, gate_w, gate_b, wbuf);

    cast_f16_kernel<<<4096, 256, 0, stream>>>(x, xh, BSD);
    cast_f16_kernel<<<4096, 256, 0, stream>>>(in_w,   iph, (long)EQ * 3 * DQ * DQ);
    cast_f16_kernel<<<4096, 256, 0, stream>>>(out_w,  oph, (long)EQ * DQ * DQ);
    cast_f16_kernel<<<4096, 256, 0, stream>>>(lin1_w, l1h, (long)EQ * DFFQ * DQ);
    cast_f16_kernel<<<4096, 256, 0, stream>>>(lin2_w, l2h, (long)EQ * DQ * DFFQ);

    const float attn_scale = 0.125f;               // 1/sqrt(hd=64)

    for (int e = 0; e < EQ; ++e) {
        // QKV projection: A[4096,1024] @ W[3072,1024]^T -> qkv f16 [B,S,3D]
        // (M, N, K) = (B*S, 3D, D)
        launch_gemm_nk(stream,
                       xh, DQ, 0, 0,
                       iph + (long)e * 3 * DQ * DQ, DQ, 0, 0,
                       nullptr, qkvh, S3D, 0, 0,
                       in_b + (long)e * 3 * DQ, 1.0f, 0,
                       BQ * SQ, S3D, DQ, 1, 1, SQ, wbuf, e);

        // Attention scores per (b,h): Q[512,64] @ K[512,64]^T * 1/8 -> f16
        launch_gemm_nk(stream,
                       qkvh, S3D, (long)SQ * S3D, 64,
                       qkvh + DQ, S3D, (long)SQ * S3D, 64,
                       nullptr, sbuf, SQ, (long)HQ * SQ * SQ, (long)SQ * SQ,
                       nullptr, attn_scale, 0,
                       SQ, SQ, 64, BQ * HQ, HQ, SQ, wbuf, e);

        softmax_rows_kernel<<<BQ * HQ * SQ, 256, 0, stream>>>(sbuf, wbuf, e);

        // Attention output per (b,h): P[512,512] @ V[512,64] -> oh f16 [B,S,D]
        {
            dim3 grid((unsigned)(SQ / 128), 1u, (unsigned)(BQ * HQ));
            gemm_wmma_kn_kernel<<<grid, 256, 0, stream>>>(
                sbuf, SQ, (long)HQ * SQ * SQ, (long)SQ * SQ,
                qkvh + 2 * DQ, S3D, (long)SQ * S3D, 64,
                oh, DQ, (long)SQ * DQ, 64,
                1.0f, SQ, HQ, wbuf, e);
        }

        // Output projection -> f32
        launch_gemm_nk(stream,
                       oh, DQ, 0, 0,
                       oph + (long)e * DQ * DQ, DQ, 0, 0,
                       oprojf, nullptr, DQ, 0, 0,
                       out_b + (long)e * DQ, 1.0f, 0,
                       BQ * SQ, DQ, DQ, 1, 1, SQ, wbuf, e);

        // LN1(x + o)
        add_ln_kernel<<<BQ * SQ, 256, 0, stream>>>(
            x, oprojf, ln1_w + (long)e * DQ, ln1_b + (long)e * DQ,
            ln1f, ln1h, wbuf, e);

        // FFN up + ReLU -> f16
        launch_gemm_nk(stream,
                       ln1h, DQ, 0, 0,
                       l1h + (long)e * DFFQ * DQ, DQ, 0, 0,
                       nullptr, ffh, DFFQ, 0, 0,
                       lin1_b + (long)e * DFFQ, 1.0f, 1,
                       BQ * SQ, DFFQ, DQ, 1, 1, SQ, wbuf, e);

        // FFN down -> f32
        launch_gemm_nk(stream,
                       ffh, DFFQ, 0, 0,
                       l2h + (long)e * DQ * DFFQ, DFFQ, 0, 0,
                       ff2f, nullptr, DQ, 0, 0,
                       lin2_b + (long)e * DQ, 1.0f, 0,
                       BQ * SQ, DQ, DFFQ, 1, 1, SQ, wbuf, e);

        // LN2(ln1 + ff)
        add_ln_kernel<<<BQ * SQ, 256, 0, stream>>>(
            ln1f, ff2f, ln2_w + (long)e * DQ, ln2_b + (long)e * DQ,
            ln2f, (half_t*)nullptr, wbuf, e);

        // out += w[b,e] * ln2
        accum_kernel<<<4096, 256, 0, stream>>>(out, ln2f, wbuf, e, BSD);
    }
}
